// FNetBlock_22058952032647
// MI455X (gfx1250) — matrix operand order repaired and checked by reading, not compiled, not run on hardware
//
#include <hip/hip_runtime.h>

typedef __attribute__((ext_vector_type(16))) _Float16     v16h;
typedef __attribute__((ext_vector_type(8)))  float        v8f;
typedef __attribute__((ext_vector_type(4)))  float        f4;
typedef __attribute__((ext_vector_type(4)))  unsigned int u4;

#define HW 65536  // 256*256 pixels per channel plane

union FragBits { v16h h; u4 q[2]; unsigned int u[8]; };

// ---------------------------------------------------------------------------
// Kernel 1: GroupNorm statistics. One block per (batch, group) = 128 blocks.
// Reduces 8 channels x 65536 px = 524288 floats -> mean, rsqrt(var+eps).
// ---------------------------------------------------------------------------
__global__ __launch_bounds__(256) void gn_stats_kernel(const float* __restrict__ x,
                                                       float* __restrict__ stats) {
  __shared__ float r1[256];
  __shared__ float r2[256];
  const int bg = blockIdx.x;                      // b*8 + g
  const f4* p4 = (const f4*)(x + (size_t)bg * 8 * HW);
  float s1 = 0.f, s2 = 0.f;
  for (int i = threadIdx.x; i < (8 * HW) / 4; i += 256) {
    f4 v = p4[i];
    s1 += v.x + v.y + v.z + v.w;
    s2 += v.x * v.x + v.y * v.y + v.z * v.z + v.w * v.w;
  }
  r1[threadIdx.x] = s1;
  r2[threadIdx.x] = s2;
  __syncthreads();
  for (int off = 128; off > 0; off >>= 1) {
    if ((int)threadIdx.x < off) {
      r1[threadIdx.x] += r1[threadIdx.x + off];
      r2[threadIdx.x] += r2[threadIdx.x + off];
    }
    __syncthreads();
  }
  if (threadIdx.x == 0) {
    const float invN = 1.f / (8.f * (float)HW);
    float mean = r1[0] * invN;
    float var  = r2[0] * invN - mean * mean;
    stats[2 * bg]     = mean;
    stats[2 * bg + 1] = rsqrtf(var + 1e-5f);
  }
}

// ---------------------------------------------------------------------------
// A-fragment (16x32 f16, M=outch, K=inch) built from f32 row-major weights.
// Layout: lanes 0-15: M=lane, V0..7 = K pairs {0..15};
//         lanes 16-31: M=lane-16, K offset +16 (matches B fragment).
// ---------------------------------------------------------------------------
__device__ inline v16h load_w_frag(const float* __restrict__ W, int mt, int kc, int lane) {
  const int m    = mt * 16 + (lane & 15);
  const int kadd = (lane >> 4) * 16;
  v16h a;
#pragma unroll
  for (int v = 0; v < 8; ++v) {
    int k = kc * 32 + 2 * v + kadd;
    a[2 * v]     = (_Float16)W[m * 64 + k];
    a[2 * v + 1] = (_Float16)W[m * 64 + k + 1];
  }
  return a;
}

// ---------------------------------------------------------------------------
// B-fragment (32x16 f16, K=channel, N=pixel) from LDS tile [px][72] (pad 72
// halves => 144B rows, 16B aligned, conflict-free). Lanes 0-15 take K 0-15,
// lanes 16-31 take K 16-31; two 16B LDS loads per fragment.
// ---------------------------------------------------------------------------
__device__ inline v16h load_b_frag(const _Float16* buf, int kc, int lane) {
  const int pl   = lane & 15;
  const int ksel = (lane >> 4) * 16;
  FragBits f;
  const u4* q = (const u4*)(buf + pl * 72 + kc * 32 + ksel);
  f.q[0] = q[0];
  f.q[1] = q[1];
  return f.h;
}

// ---------------------------------------------------------------------------
// Kernel 2: fused  xn = (x-mean)*rstd*gw+gb ;  h = relu(W1 xn + b1)
//                  out = xn + W2 h + b2
// (FFT->IFFT round trip is the identity and is elided.)
// Each wave: 4 tiles of 16 pixels. Block = 8 waves = 512 px. Grid = 2048.
// ---------------------------------------------------------------------------
__global__ __launch_bounds__(256) void fnet_fused_kernel(
    const float* __restrict__ x,  const float* __restrict__ gw,
    const float* __restrict__ gb, const float* __restrict__ w1,
    const float* __restrict__ b1, const float* __restrict__ w2,
    const float* __restrict__ b2, const float* __restrict__ stats,
    float* __restrict__ out) {
  __shared__ float s_scale[64], s_shift[64], s_b1[64], s_b2[64];
  __shared__ float s_xnf[8][16 * 68];                                   // f32 residual copy
  __shared__ __attribute__((aligned(16))) _Float16 s_xh[8][16 * 72];    // f16 act / h tile

  const int b      = blockIdx.x >> 7;           // 16 batches x 128 blocks
  const int blk_px = (blockIdx.x & 127) * 512;
  const int wave   = threadIdx.x >> 5;
  const int lane   = threadIdx.x & 31;

  if (threadIdx.x < 64) {
    const int c  = threadIdx.x;
    const int sg = 2 * (b * 8 + (c >> 3));
    float mean = stats[sg], rstd = stats[sg + 1];
    float sc   = rstd * gw[c];
    s_scale[c] = sc;
    s_shift[c] = gb[c] - mean * sc;
    s_b1[c]    = b1[c];
    s_b2[c]    = b2[c];
  }

  // Both weight matrices as f16 A-fragments, resident in registers.
  v16h w1f[4][2], w2f[4][2];
#pragma unroll
  for (int mt = 0; mt < 4; ++mt)
#pragma unroll
    for (int kc = 0; kc < 2; ++kc) {
      w1f[mt][kc] = load_w_frag(w1, mt, kc, lane);
      w2f[mt][kc] = load_w_frag(w2, mt, kc, lane);
    }
  __syncthreads();

  const int pl   = lane & 15;
  const int hsel = lane >> 4;
  float*     xnrow = &s_xnf[wave][0];
  _Float16*  xhrow = &s_xh[wave][0];
  const size_t img_base = (size_t)b * 64 * HW;

  for (int t = 0; t < 4; ++t) {
    const int px0 = blk_px + wave * 64 + t * 16;
    const float* xb = x + img_base + px0;

    // ---- stage: normalize 64ch x 16px into LDS (f32 + f16 [px][ch]) ----
#pragma unroll
    for (int it = 0; it < 8; ++it) {
      const int c = it * 8 + (lane >> 2);
      const int p = (lane & 3) * 4;
      f4 xv = *(const f4*)(xb + (size_t)c * HW + p);
      const float sc = s_scale[c], sh = s_shift[c];
#pragma unroll
      for (int j = 0; j < 4; ++j) {
        float xn = xv[j] * sc + sh;
        xnrow[(p + j) * 68 + c] = xn;
        xhrow[(p + j) * 72 + c] = (_Float16)xn;
      }
    }
    __syncthreads();

    // ---- GEMM1: acc = W1 * xn ----
    v16h bf0 = load_b_frag(xhrow, 0, lane);
    v16h bf1 = load_b_frag(xhrow, 1, lane);
    v8f acc[4];
#pragma unroll
    for (int mt = 0; mt < 4; ++mt) {
      v8f c0 = {};
      c0 = __builtin_amdgcn_wmma_f32_16x16x32_f16(false, w1f[mt][0], false, bf0,
                                                  (short)0, c0, false, false);
      c0 = __builtin_amdgcn_wmma_f32_16x16x32_f16(false, w1f[mt][1], false, bf1,
                                                  (short)0, c0, false, false);
      acc[mt] = c0;
    }

    // ---- bias + ReLU -> f16 h tile (reuse xh buffer; wave-local, in-order) ----
#pragma unroll
    for (int mt = 0; mt < 4; ++mt)
#pragma unroll
      for (int r = 0; r < 8; ++r) {
        const int oc = mt * 16 + r + hsel * 8;
        float hv = acc[mt][r] + s_b1[oc];
        hv = hv > 0.f ? hv : 0.f;
        xhrow[pl * 72 + oc] = (_Float16)hv;
      }
    __syncthreads();

    // ---- GEMM2: acc = W2 * h ----
    v16h hf0 = load_b_frag(xhrow, 0, lane);
    v16h hf1 = load_b_frag(xhrow, 1, lane);
#pragma unroll
    for (int mt = 0; mt < 4; ++mt) {
      v8f c0 = {};
      c0 = __builtin_amdgcn_wmma_f32_16x16x32_f16(false, w2f[mt][0], false, hf0,
                                                  (short)0, c0, false, false);
      c0 = __builtin_amdgcn_wmma_f32_16x16x32_f16(false, w2f[mt][1], false, hf1,
                                                  (short)0, c0, false, false);
      acc[mt] = c0;
    }

    // ---- out = xn(f32) + acc + b2 ; 64B-contiguous half-wave stores ----
    float* ob = out + img_base + px0;
#pragma unroll
    for (int mt = 0; mt < 4; ++mt)
#pragma unroll
      for (int r = 0; r < 8; ++r) {
        const int oc = mt * 16 + r + hsel * 8;
        ob[(size_t)oc * HW + pl] = acc[mt][r] + s_b2[oc] + xnrow[pl * 68 + oc];
      }
    __syncthreads();
  }
}

extern "C" void kernel_launch(void* const* d_in, const int* in_sizes, int n_in,
                              void* d_out, int out_size, void* d_ws, size_t ws_size,
                              hipStream_t stream) {
  const float* x  = (const float*)d_in[0];
  const float* gw = (const float*)d_in[1];
  const float* gb = (const float*)d_in[2];
  const float* w1 = (const float*)d_in[3];
  const float* b1 = (const float*)d_in[4];
  const float* w2 = (const float*)d_in[5];
  const float* b2 = (const float*)d_in[6];
  float* out   = (float*)d_out;
  float* stats = (float*)d_ws;   // 128 * 2 floats

  gn_stats_kernel<<<128, 256, 0, stream>>>(x, stats);
  fnet_fused_kernel<<<2048, 256, 0, stream>>>(x, gw, gb, w1, b1, w2, b2, stats, out);
}
